// GraphAttentionLayer_22127671509810
// MI455X (gfx1250) — matrix operand order repaired and checked by reading, not compiled, not run on hardware
//
#include <hip/hip_runtime.h>
#include <hip/hip_bf16.h>
#include <cmath>

// ---------------- types ----------------
typedef _Float16 v16h __attribute__((ext_vector_type(16)));
typedef float    v8f  __attribute__((ext_vector_type(8)));
typedef int      v4i  __attribute__((ext_vector_type(4)));
typedef float    v4f  __attribute__((ext_vector_type(4)));

#define GAT_N     8192
#define GAT_INF   128
#define GAT_OUTF  64
#define GAT_ALPHA 0.2f
#define GAT_LOG2E 1.44269504088896f

// =====================================================================
// Kernel 1: h = input @ W  (store transposed f16), f1 = h@a1, f2 = h@a2
// one block (64 threads) per row
// =====================================================================
__global__ __launch_bounds__(64) void gat_h_kernel(
    const float* __restrict__ inp, const float* __restrict__ W,
    const float* __restrict__ a, _Float16* __restrict__ hT,
    float* __restrict__ f1, float* __restrict__ f2)
{
    const int row = blockIdx.x;
    const int col = threadIdx.x;           // 0..63
    const float* ir = inp + (size_t)row * GAT_INF;

    float acc = 0.0f;
#pragma unroll 8
    for (int k = 0; k < GAT_INF; ++k)
        acc = fmaf(ir[k], W[k * GAT_OUTF + col], acc);

    // transposed f16 copy for WMMA B-operand tiles
    hT[(size_t)col * GAT_N + row] = (_Float16)acc;

    __shared__ float s1[64];
    __shared__ float s2[64];
    s1[col] = acc * a[col];
    s2[col] = acc * a[GAT_OUTF + col];
    __syncthreads();
#pragma unroll
    for (int off = 32; off > 0; off >>= 1) {
        if (col < off) { s1[col] += s1[col + off]; s2[col] += s2[col + off]; }
        __syncthreads();
    }
    if (col == 0) { f1[row] = s1[0]; f2[row] = s2[0]; }
}

// =====================================================================
// Kernel 1.5: f2max = max(f2)  (exact upper-bound shift for softmax)
// =====================================================================
__global__ __launch_bounds__(256) void gat_f2max_kernel(
    const float* __restrict__ f2, float* __restrict__ f2max)
{
    __shared__ float red[256];
    const int tid = threadIdx.x;
    float v = -3.0e38f;
    for (int i = tid; i < GAT_N; i += 256) v = fmaxf(v, f2[i]);
    red[tid] = v;
    __syncthreads();
#pragma unroll
    for (int off = 128; off > 0; off >>= 1) {
        if (tid < off) red[tid] = fmaxf(red[tid], red[tid + off]);
        __syncthreads();
    }
    if (tid == 0) *f2max = red[0];
}

// =====================================================================
// Kernel 2: fused mask + fixed-shift softmax + (attn @ h) + ELU
// grid = 512 blocks (one 16-row tile each), 256 threads = 8 waves,
// each wave handles 2048 columns from 2048*wave (past N for waves 4..7), partials summed via LDS.
// M_i = leaky_relu(f1_i + max(f2)) >= every score in row i, so
// p = exp(s - M_i) needs no running max / rescaling.
// =====================================================================
__global__ __launch_bounds__(256) void gat_attn_kernel(
    const int* __restrict__ adj, const _Float16* __restrict__ hT,
    const float* __restrict__ f1, const float* __restrict__ f2,
    const float* __restrict__ f2maxp, float* __restrict__ out)
{
    const int lane  = threadIdx.x & 31;
    const int wave  = threadIdx.x >> 5;        // 0..7 : 2048-column slice
    const int r0    = blockIdx.x * 16;         // row tile base
    const int rlane = lane & 15;               // row within tile (A layout)
    const int kbase = (lane < 16) ? 0 : 16;    // A-layout K offset inside 32-tile
    const int koffB = (lane < 16) ? 0 : 16;    // B-layout K offset inside 32-tile
    const int rbase = (lane < 16) ? 0 : 8;     // C-layout row base

    const int row   = r0 + rlane;
    const float f1v = f1[row];
    const float f2m = *f2maxp;
    // exact row upper bound of scores (leaky_relu is monotone)
    const float tM  = f1v + f2m;
    const float M   = fmaxf(tM, GAT_ALPHA * tM);
    const float C   = -M * GAT_LOG2E;          // p = exp2(t*log2e + C)

    float lsum = 0.0f;                         // lane-local partial row sum
    v8f acc0 = {}; v8f acc1 = {}; v8f acc2 = {}; v8f acc3 = {};

    const int* adjRow = adj + (size_t)row * GAT_N;
    const int jbeg = wave * (GAT_N / 4);
    const int jend = jbeg + (GAT_N / 4);

    for (int j0 = jbeg; j0 < jend; j0 += 32) {
        // ---- stream adj tile (read-once: non-temporal) + f2 slice,
        //      directly in contiguous-K A order (k = 16*(lane/16) + e) ----
        const int* ap = adjRow + j0 + kbase;
        __builtin_prefetch(adjRow + j0 + 32 + kbase, 0, 0);
        v4i A0 = __builtin_nontemporal_load((const v4i*)(ap + 0));
        v4i A1 = __builtin_nontemporal_load((const v4i*)(ap + 4));
        v4i A2 = __builtin_nontemporal_load((const v4i*)(ap + 8));
        v4i A3 = __builtin_nontemporal_load((const v4i*)(ap + 12));
        const float* gp = f2 + j0 + kbase;
        v4f G0 = *(const v4f*)(gp + 0);
        v4f G1 = *(const v4f*)(gp + 4);
        v4f G2 = *(const v4f*)(gp + 8);
        v4f G3 = *(const v4f*)(gp + 12);

        int   av[16]; float gv[16];
#pragma unroll
        for (int e = 0; e < 4; ++e) {
            av[e]      = A0[e]; gv[e]      = G0[e];
            av[4 + e]  = A1[e]; gv[4 + e]  = G1[e];
            av[8 + e]  = A2[e]; gv[8 + e]  = G2[e];
            av[12 + e] = A3[e]; gv[12 + e] = G3[e];
        }

        // ---- p = adj ? exp(leaky_relu(f1+f2) - M) : 0  -> f16 A operand ----
        v16h pA;
#pragma unroll
        for (int e = 0; e < 16; ++e) {
            float t = f1v + gv[e];
            t = fmaxf(t, GAT_ALPHA * t);                 // leaky_relu
            float p = exp2f(fmaf(t, GAT_LOG2E, C));      // exp(t - M)
            p = (av[e] > 0) ? p : 0.0f;                  // adjacency mask
            lsum += p;
            pA[e] = (_Float16)p;
        }

        // ---- B tiles from hT (contiguous 32B per lane) + WMMA ----
        const size_t bofs = (size_t)rlane * GAT_N + (size_t)(j0 + koffB);
        v16h b0 = *(const v16h*)(hT + bofs);
        v16h b1 = *(const v16h*)(hT + bofs + (size_t)16 * GAT_N);
        v16h b2 = *(const v16h*)(hT + bofs + (size_t)32 * GAT_N);
        v16h b3 = *(const v16h*)(hT + bofs + (size_t)48 * GAT_N);
        acc0 = __builtin_amdgcn_wmma_f32_16x16x32_f16(false, pA, false, b0, (short)0, acc0, false, false);
        acc1 = __builtin_amdgcn_wmma_f32_16x16x32_f16(false, pA, false, b1, (short)0, acc1, false, false);
        acc2 = __builtin_amdgcn_wmma_f32_16x16x32_f16(false, pA, false, b2, (short)0, acc2, false, false);
        acc3 = __builtin_amdgcn_wmma_f32_16x16x32_f16(false, pA, false, b3, (short)0, acc3, false, false);
    }

    // row sum: lanes L and L+16 hold disjoint K-halves of row L&15
    lsum += __shfl_xor(lsum, 16, 32);

    // ---- sum the 4 column-quarter partials via LDS (same shift M) ----
    __shared__ float sacc[4 * 16 * GAT_OUTF];   // 16 KB
    __shared__ float sl[4 * 16];

#pragma unroll
    for (int r = 0; r < 8; ++r) {
        const int rr = rbase + r;
        sacc[(wave * 16 + rr) * GAT_OUTF +  0 + rlane] = acc0[r];
        sacc[(wave * 16 + rr) * GAT_OUTF + 16 + rlane] = acc1[r];
        sacc[(wave * 16 + rr) * GAT_OUTF + 32 + rlane] = acc2[r];
        sacc[(wave * 16 + rr) * GAT_OUTF + 48 + rlane] = acc3[r];
    }
    if (lane < 16) sl[wave * 16 + rlane] = lsum;
    __syncthreads();

    for (int idx = threadIdx.x; idx < 16 * GAT_OUTF; idx += 256) {
        const int rr = idx >> 6;        // 0..15
        const int cc = idx & 63;        // 0..63
        float lg = sl[rr] + sl[16 + rr] + sl[32 + rr] + sl[48 + rr];
        float v  = sacc[(0 * 16 + rr) * GAT_OUTF + cc]
                 + sacc[(1 * 16 + rr) * GAT_OUTF + cc]
                 + sacc[(2 * 16 + rr) * GAT_OUTF + cc]
                 + sacc[(3 * 16 + rr) * GAT_OUTF + cc];
        v = v / lg;
        v = (v > 0.0f) ? v : expm1f(v);           // ELU
        out[(size_t)(r0 + rr) * GAT_OUTF + cc] = v;
    }
}

// =====================================================================
extern "C" void kernel_launch(void* const* d_in, const int* in_sizes, int n_in,
                              void* d_out, int out_size, void* d_ws, size_t ws_size,
                              hipStream_t stream) {
    const float* inp = (const float*)d_in[0];   // (8192,128) f32
    const int*   adj = (const int*)  d_in[1];   // (8192,8192) i32
    const float* W   = (const float*)d_in[2];   // (128,64) f32
    const float* a   = (const float*)d_in[3];   // (128,1) f32
    float* out = (float*)d_out;                 // (8192,64) f32

    char* ws = (char*)d_ws;
    _Float16* hT = (_Float16*)ws;                                   // 1 MB
    float* f1    = (float*)(ws + (size_t)GAT_OUTF * GAT_N * sizeof(_Float16));
    float* f2    = f1 + GAT_N;
    float* f2max = f2 + GAT_N;

    gat_h_kernel<<<GAT_N, 64, 0, stream>>>(inp, W, a, hT, f1, f2);
    gat_f2max_kernel<<<1, 256, 0, stream>>>(f2, f2max);
    gat_attn_kernel<<<GAT_N / 16, 256, 0, stream>>>(adj, hT, f1, f2, f2max, out);
}
